// ClusterAttention_8993661518302
// MI455X (gfx1250) — matrix operand order repaired and checked by reading, not compiled, not run on hardware
//
#include <hip/hip_runtime.h>
#include <hip/hip_bf16.h>

// ---------------------------------------------------------------------------
// ClusterAttention for MI455X (gfx1250), wave32 + WMMA.
// Heavy GEMMs: f16 operands, f32 accumulation via v_wmma_f32_16x16x32_f16,
// 2x2 register blocking (32x32 C per wave) for 1:2 wmma:b128-load ratio.
// Shapes: B=4, N=16384 (2^14), D=256, H=8, M=32, HD=32.
// ---------------------------------------------------------------------------

typedef __attribute__((ext_vector_type(16))) _Float16 v16h;
typedef __attribute__((ext_vector_type(8)))  _Float16 h8;
typedef __attribute__((ext_vector_type(8)))  float    v8f;

#define BB   4
#define NN   16384
#define LOGN 14
#define DD   256
#define HH   8
#define MM   32
#define HD   32
#define BN   (BB * NN)   // 65536 rows

// ---- WMMA fragment loaders (layouts per CDNA5 ISA 7.12.2) ------
// A (16x32 f16): lane&15 = row; lanes 0-15 hold K {0..7,16..23}, lanes 16-31
// hold K {8..15,24..31}  -> two contiguous 16-byte loads per lane.
__device__ __forceinline__ v16h frag_a(const _Float16* p0, int ld, int lane) {
  const _Float16* q = p0 + (long)(lane & 15) * ld + ((lane >> 4) << 3);
  h8 lo = *(const h8*)(q);
  h8 hi = *(const h8*)(q + 16);
  v16h f;
#pragma unroll
  for (int i = 0; i < 8; ++i) { f[i] = lo[i]; f[i + 8] = hi[i]; }
  return f;
}
// B (32x16 f16): lane&15 = column; lanes 0-15 hold K {0..7,16..23}, lanes 16-31
// hold K {8..15,24..31} (read from row-major [col][K]).
__device__ __forceinline__ v16h frag_b(const _Float16* p0, int ld, int lane) {
  const _Float16* q = p0 + (long)(lane & 15) * ld + ((lane >> 4) << 3);
  h8 lo = *(const h8*)(q);
  h8 hi = *(const h8*)(q + 16);
  v16h f;
#pragma unroll
  for (int i = 0; i < 8; ++i) { f[i] = lo[i]; f[i + 8] = hi[i]; }
  return f;
}
__device__ __forceinline__ v8f wmma_f16(v16h a, v16h b, v8f c) {
  return __builtin_amdgcn_wmma_f32_16x16x32_f16(false, a, false, b,
                                                (short)0, c, false, false);
}
// C/D tile element (i of v8f): m = i + 8*(lane>=16), n = lane&15.

// ---------------------------------------------------------------------------
// 0) fp32 -> f16 conversion (x and all GEMM weights)
__global__ void k_cvt(const float* __restrict__ in, _Float16* __restrict__ out,
                      long n) {
  long i = (long)blockIdx.x * blockDim.x + threadIdx.x;
  const long stride = (long)gridDim.x * blockDim.x;
  for (; i < n; i += stride) out[i] = (_Float16)in[i];
}

// ---------------------------------------------------------------------------
// 1) kv = x @ kv_w^T + kv_b.  Each wave: 32x32 C tile (2x2 blocking).
//    k-half -> kh[bn][c] (K-contiguous for scores),
//    v-half -> xvT[b][h][hd][n] (K=n contiguous for the ct GEMM).
__global__ void k_gemm_kv(const _Float16* __restrict__ xh,
                          const _Float16* __restrict__ wh,
                          const float* __restrict__ bias,
                          _Float16* __restrict__ kh,
                          _Float16* __restrict__ xvT) {
  const int lane = threadIdx.x & 31, wave = threadIdx.x >> 5;
  const int row0 = blockIdx.x * 32;                 // 2048 row groups
  const int c0 = (blockIdx.y * 4 + wave) * 32;      // 16 col groups
  const _Float16* pa = xh + (long)row0 * DD;
  const _Float16* pb = wh + (long)c0 * DD;
  v8f acc[2][2] = {};
#pragma unroll
  for (int k = 0; k < DD; k += 32) {
    const v16h a0 = frag_a(pa + k, DD, lane);
    const v16h a1 = frag_a(pa + 16 * DD + k, DD, lane);
    const v16h b0 = frag_b(pb + k, DD, lane);
    const v16h b1 = frag_b(pb + 16 * DD + k, DD, lane);
    acc[0][0] = wmma_f16(a0, b0, acc[0][0]);
    acc[0][1] = wmma_f16(a0, b1, acc[0][1]);
    acc[1][0] = wmma_f16(a1, b0, acc[1][0]);
    acc[1][1] = wmma_f16(a1, b1, acc[1][1]);
  }
  const int mb = (lane >> 4) << 3;
#pragma unroll
  for (int r = 0; r < 2; ++r) {
#pragma unroll
    for (int cc = 0; cc < 2; ++cc) {
      const int c = c0 + cc * 16 + (lane & 15);
      const float bv = bias[c];
#pragma unroll
      for (int i = 0; i < 8; ++i) {
        const long bn = row0 + r * 16 + mb + i;
        const float v = acc[r][cc][i] + bv;
        if (c < DD) {
          kh[bn * DD + c] = (_Float16)v;
        } else {
          const int hd_ = c - DD;
          const long b = bn >> LOGN, n = bn & (NN - 1);
          xvT[(((b * HH + (hd_ >> 5)) * HD + (hd_ & 31)) << LOGN) + n] =
              (_Float16)v;
        }
      }
    }
  }
}

// ---------------------------------------------------------------------------
// 2) scores[b,h,m,n] = wtq[h] @ xk[b,h]^T, stored transposed as
//    scoresT[bn][h*32+m].  K=HD=32.  A fragment reused across 4 n-tiles.
__global__ void k_scores(const _Float16* __restrict__ wtqh,
                         const _Float16* __restrict__ kh,
                         _Float16* __restrict__ scoresT) {
  const int lane = threadIdx.x & 31, wave = threadIdx.x >> 5;
  const int job = blockIdx.x * 4 + wave;       // B*H*2*256 jobs
  const int ntg = job & 255, mt = (job >> 8) & 1;
  const int h = (job >> 9) & 7, b = job >> 12;
  const v16h a = frag_a(wtqh + (h * MM + mt * 16) * HD, HD, lane);
  const int m0 = mt * 16 + ((lane >> 4) << 3);
#pragma unroll
  for (int j = 0; j < 4; ++j) {
    const int nt = ntg * 4 + j;
    const _Float16* pb = kh + ((long)(b << LOGN) + nt * 16) * DD + h * HD;
    v8f acc = {};
    acc = wmma_f16(a, frag_b(pb, DD, lane), acc);
    const long bn = (long)(b << LOGN) + nt * 16 + (lane & 15);
    h8 o;
#pragma unroll
    for (int i = 0; i < 8; ++i) o[i] = (_Float16)acc[i];
    *(h8*)(scoresT + bn * DD + h * MM + m0) = o;   // 8 contiguous halves
  }
}

// ---------------------------------------------------------------------------
// 3) mix: scores2[bn][o] = mix_w[o,:] . scoresT[bn][:]   (K=256, 2x2 blocked)
__global__ void k_mix(const _Float16* __restrict__ mixh,
                      const _Float16* __restrict__ scoresT,
                      float* __restrict__ scores2) {
  const int lane = threadIdx.x & 31, wave = threadIdx.x >> 5;
  const int job = blockIdx.x * 4 + wave;       // B*8*512 jobs
  const int nt = job & 511, ot = (job >> 9) & 7, b = job >> 12;
  const long bn0 = (long)(b << LOGN) + nt * 32;
  const _Float16* pa = mixh + (long)ot * 32 * DD;   // rows = o
  const _Float16* pb = scoresT + bn0 * DD;          // cols = n
  v8f acc[2][2] = {};
#pragma unroll
  for (int k = 0; k < DD; k += 32) {
    const v16h a0 = frag_a(pa + k, DD, lane);
    const v16h a1 = frag_a(pa + 16 * DD + k, DD, lane);
    const v16h b0 = frag_b(pb + k, DD, lane);
    const v16h b1 = frag_b(pb + 16 * DD + k, DD, lane);
    acc[0][0] = wmma_f16(a0, b0, acc[0][0]);
    acc[0][1] = wmma_f16(a0, b1, acc[0][1]);
    acc[1][0] = wmma_f16(a1, b0, acc[1][0]);
    acc[1][1] = wmma_f16(a1, b1, acc[1][1]);
  }
#pragma unroll
  for (int r = 0; r < 2; ++r) {
#pragma unroll
    for (int cc = 0; cc < 2; ++cc) {
      const long n = bn0 + cc * 16 + (lane & 15);
      float* dst = scores2 + n * DD + ot * 32 + r * 16 + ((lane >> 4) << 3);
#pragma unroll
      for (int i = 0; i < 8; ++i) dst[i] = acc[r][cc][i];
    }
  }
}

// ---------------------------------------------------------------------------
// 4) softmax over the M=32 cluster dim (contiguous per (b,n,h)).
//    Emit cw in both layouts: cwT[bn][h*32+m] (f16) and cw_mn[(bh m)][n] (f16).
__global__ void k_softmax(const float* __restrict__ scores2,
                          _Float16* __restrict__ cwT,
                          _Float16* __restrict__ cw_mn) {
  const long t = (long)blockIdx.x * blockDim.x + threadIdx.x;  // B*N*H
  const int h = (int)(t & 7);
  const long bn = t >> 3;
  const long b = bn >> LOGN, n = bn & (NN - 1);
  const float* src = scores2 + bn * DD + h * MM;
  float v[MM], mx = -1e30f;
#pragma unroll
  for (int m = 0; m < MM; ++m) { v[m] = src[m]; mx = fmaxf(mx, v[m]); }
  float s = 0.f;
#pragma unroll
  for (int m = 0; m < MM; ++m) { v[m] = __expf(v[m] - mx); s += v[m]; }
  const float inv = 1.f / s;
  _Float16* d1 = cwT + bn * DD + h * MM;
#pragma unroll
  for (int m = 0; m < MM; ++m) {
    const float w = v[m] * inv;
    d1[m] = (_Float16)w;
    cw_mn[(((b * HH + h) * MM + m) << LOGN) + n] = (_Float16)w;
  }
}

// ---------------------------------------------------------------------------
// 4b) rs[b,h,m] = sum_n cw[b,h,m,n]
__global__ void k_rowsum(const _Float16* __restrict__ cw_mn,
                         float* __restrict__ rs) {
  __shared__ float red[256];
  const _Float16* p = cw_mn + ((long)blockIdx.x << LOGN);
  float s = 0.f;
  for (int n = threadIdx.x; n < NN; n += 256) s += (float)p[n];
  red[threadIdx.x] = s;
  __syncthreads();
  for (int st = 128; st > 0; st >>= 1) {
    if ((int)threadIdx.x < st) red[threadIdx.x] += red[threadIdx.x + st];
    __syncthreads();
  }
  if (threadIdx.x == 0) rs[blockIdx.x] = red[0];
}

// ---------------------------------------------------------------------------
// 5) ct[b,h,m,d] = sum_n cw[m,n] * xv[n,d].  Split-K: 16 chunks of 1024 over
//    N, deterministic per-chunk partials (no float atomics), then reduce.
#define CTCH 16
#define CTSZ (NN / CTCH)   // 1024
__global__ void k_ct(const _Float16* __restrict__ cw_mn,
                     const _Float16* __restrict__ xvT,
                     float* __restrict__ ctP) {
  const int lane = threadIdx.x & 31, wave = threadIdx.x >> 5;
  const int chunk = blockIdx.x & (CTCH - 1);
  const int bh = blockIdx.x >> 4;
  const int b = bh >> 3, h = bh & 7;
  const int mi = wave >> 1, di = wave & 1;
  const _Float16* pa =
      cw_mn + ((long)(bh * MM + mi * 16) << LOGN) + chunk * CTSZ;
  const _Float16* pb =
      xvT + ((long)(bh * HD + di * 16) << LOGN) + chunk * CTSZ;
  v8f acc = {};
  for (int n = 0; n < CTSZ; n += 32) {
    __builtin_prefetch(pa + n + 512, 0, 1);
    __builtin_prefetch(pb + n + 512, 0, 1);
    acc = wmma_f16(frag_a(pa + n, NN, lane), frag_b(pb + n, NN, lane), acc);
  }
  const int m = mi * 16 + ((lane >> 4) << 3);
  const int d = di * 16 + (lane & 15);
#pragma unroll
  for (int i = 0; i < 8; ++i)
    ctP[(long)chunk * (BB * MM * DD) + (long)(b * MM + m + i) * DD + h * HD +
        d] = acc[i];
}

__global__ void k_ctred(const float* __restrict__ ctP,
                        float* __restrict__ ctBMD) {
  const int idx = blockIdx.x * 256 + threadIdx.x;   // BB*MM*DD = 32768
  float s = 0.f;
#pragma unroll
  for (int c = 0; c < CTCH; ++c) s += ctP[(long)c * (BB * MM * DD) + idx];
  ctBMD[idx] = s;
}

// ---------------------------------------------------------------------------
// 6a) ct /= (rowsum + 1e-5); LayerNorm1 over D.  One block per (b,m).
__global__ void k_norm_ln1(const float* __restrict__ ctBMD,
                           const float* __restrict__ rs,
                           const float* __restrict__ g,
                           const float* __restrict__ be,
                           float* __restrict__ ct_ln) {
  __shared__ float red[256];
  const int b = blockIdx.x >> 5, m = blockIdx.x & 31;
  const int t = threadIdx.x;
  const float v = ctBMD[(long)blockIdx.x * DD + t] /
                  (rs[(b * HH + (t >> 5)) * MM + m] + 1e-5f);
  red[t] = v;
  __syncthreads();
  for (int st = 128; st > 0; st >>= 1) {
    if (t < st) red[t] += red[t + st];
    __syncthreads();
  }
  const float mu = red[0] * (1.f / 256.f);
  __syncthreads();
  red[t] = (v - mu) * (v - mu);
  __syncthreads();
  for (int st = 128; st > 0; st >>= 1) {
    if (t < st) red[t] += red[t + st];
    __syncthreads();
  }
  const float var = red[0] * (1.f / 256.f);
  ct_ln[(long)blockIdx.x * DD + t] =
      (v - mu) * rsqrtf(var + 1e-5f) * g[t] + be[t];
}

// ---------------------------------------------------------------------------
// 6b) tiny MHA over M=32 tokens (scalar fp32; ~0.02 GFLOP total)
__global__ void k_qkv(const float* __restrict__ ct_ln,
                      const float* __restrict__ w,
                      const float* __restrict__ bias,
                      float* __restrict__ qkv) {
  const long idx = (long)blockIdx.x * 256 + threadIdx.x;  // B*32*768
  const int o = (int)(idx % 768);
  const long bm = idx / 768;
  const float* xr = ct_ln + bm * DD;
  const float* wr = w + (long)o * DD;
  float s = bias[o];
  for (int i = 0; i < DD; ++i) s += xr[i] * wr[i];
  qkv[idx] = s;
}

__global__ void k_att(const float* __restrict__ qkv, float* __restrict__ att) {
  const int b = blockIdx.x >> 3, h = blockIdx.x & 7;
  const int mq = threadIdx.x;  // 32 threads = 1 wave
  const float* q = qkv + (long)(b * MM + mq) * 768 + h * HD;
  const float scale = 0.1767766952966369f;  // 1/sqrt(32)
  float s[MM], mx = -1e30f;
  for (int mk = 0; mk < MM; ++mk) {
    const float* kk = qkv + (long)(b * MM + mk) * 768 + DD + h * HD;
    float d = 0.f;
    for (int i = 0; i < HD; ++i) d += q[i] * kk[i];
    s[mk] = d * scale;
    mx = fmaxf(mx, s[mk]);
  }
  float sum = 0.f;
  for (int mk = 0; mk < MM; ++mk) { s[mk] = __expf(s[mk] - mx); sum += s[mk]; }
  const float inv = 1.f / sum;
  float* dst = att + (long)((b * HH + h) * MM + mq) * MM;
  for (int mk = 0; mk < MM; ++mk) dst[mk] = s[mk] * inv;
}

__global__ void k_mo(const float* __restrict__ att,
                     const float* __restrict__ qkv, float* __restrict__ mo) {
  const long idx = (long)blockIdx.x * 256 + threadIdx.x;  // B*32*256
  const int o = (int)(idx & 255);
  const long bm = idx >> 8;
  const int b = (int)(bm >> 5), m = (int)(bm & 31);
  const float* a = att + (long)((b * HH + (o >> 5)) * MM + m) * MM;
  float s = 0.f;
  for (int k = 0; k < MM; ++k)
    s += a[k] * qkv[(long)(b * MM + k) * 768 + 512 + o];
  mo[idx] = s;
}

// mo @ mo_w^T + mo_b; residual ct_ln*alphaC; LayerNorm2; write otT[b,h,hd,m]
__global__ void k_moproj_ln2(const float* __restrict__ mo,
                             const float* __restrict__ w,
                             const float* __restrict__ bias,
                             const float* __restrict__ ct_ln,
                             const float* __restrict__ alphaC,
                             const float* __restrict__ g,
                             const float* __restrict__ be,
                             _Float16* __restrict__ otT) {
  __shared__ float red[256];
  const int bm = blockIdx.x;
  const int b = bm >> 5, m = bm & 31;
  const int o = threadIdx.x;
  const float* xr = mo + (long)bm * DD;
  const float* wr = w + (long)o * DD;
  float s = bias[o];
  for (int i = 0; i < DD; ++i) s += xr[i] * wr[i];
  const float r = ct_ln[(long)bm * DD + o] * alphaC[o] + s;
  red[o] = r;
  __syncthreads();
  for (int st = 128; st > 0; st >>= 1) {
    if (o < st) red[o] += red[o + st];
    __syncthreads();
  }
  const float mu = red[0] * (1.f / 256.f);
  __syncthreads();
  red[o] = (r - mu) * (r - mu);
  __syncthreads();
  for (int st = 128; st > 0; st >>= 1) {
    if (o < st) red[o] += red[o + st];
    __syncthreads();
  }
  const float var = red[0] * (1.f / 256.f);
  const float y = (r - mu) * rsqrtf(var + 1e-5f) * g[o] + be[o];
  otT[(long)((b * HH + (o >> 5)) * HD + (o & 31)) * MM + m] = (_Float16)y;
}

// ---------------------------------------------------------------------------
// 7) broadcast: out1[b,n,h*32+hd] = sum_m cwT[b,n,h,m] * ot[b,h,m,hd]  (K=32)
//    B fragment (otT) reused across 4 n-tiles.
__global__ void k_bcast(const _Float16* __restrict__ cwT,
                        const _Float16* __restrict__ otT,
                        _Float16* __restrict__ out1) {
  const int lane = threadIdx.x & 31, wave = threadIdx.x >> 5;
  const int job = blockIdx.x * 4 + wave;      // B*H*2*256 jobs
  const int ntg = job & 255, hdt = (job >> 8) & 1;
  const int h = (job >> 9) & 7, b = job >> 12;
  const v16h bf =
      frag_b(otT + (long)((b * HH + h) * HD + hdt * 16) * MM, MM, lane);
  const int hd = hdt * 16 + (lane & 15);
#pragma unroll
  for (int j = 0; j < 4; ++j) {
    const int nt = ntg * 4 + j;
    const _Float16* pa = cwT + ((long)(b << LOGN) + nt * 16) * DD + h * MM;
    v8f acc = {};
    acc = wmma_f16(frag_a(pa, DD, lane), bf, acc);
    const long base =
        ((long)(b << LOGN) + nt * 16 + ((lane >> 4) << 3)) * DD + h * HD + hd;
#pragma unroll
    for (int i = 0; i < 8; ++i) out1[base + (long)i * DD] = (_Float16)acc[i];
  }
}

// ---------------------------------------------------------------------------
// 8) final projection: out = out1 @ out_w^T + out_b  (fp32 output, 2x2 tile)
__global__ void k_out(const _Float16* __restrict__ out1,
                      const _Float16* __restrict__ wh,
                      const float* __restrict__ bias,
                      float* __restrict__ out) {
  const int lane = threadIdx.x & 31, wave = threadIdx.x >> 5;
  const int row0 = blockIdx.x * 32;                 // 2048 row groups
  const int c0 = (blockIdx.y * 4 + wave) * 32;      // 8 col groups
  const _Float16* pa = out1 + (long)row0 * DD;
  const _Float16* pb = wh + (long)c0 * DD;
  v8f acc[2][2] = {};
#pragma unroll
  for (int k = 0; k < DD; k += 32) {
    const v16h a0 = frag_a(pa + k, DD, lane);
    const v16h a1 = frag_a(pa + 16 * DD + k, DD, lane);
    const v16h b0 = frag_b(pb + k, DD, lane);
    const v16h b1 = frag_b(pb + 16 * DD + k, DD, lane);
    acc[0][0] = wmma_f16(a0, b0, acc[0][0]);
    acc[0][1] = wmma_f16(a0, b1, acc[0][1]);
    acc[1][0] = wmma_f16(a1, b0, acc[1][0]);
    acc[1][1] = wmma_f16(a1, b1, acc[1][1]);
  }
  const int mb = (lane >> 4) << 3;
#pragma unroll
  for (int r = 0; r < 2; ++r) {
#pragma unroll
    for (int cc = 0; cc < 2; ++cc) {
      const int c = c0 + cc * 16 + (lane & 15);
      const float bv = bias[c];
      const long base = ((long)row0 + r * 16 + mb) * DD + c;
#pragma unroll
      for (int i = 0; i < 8; ++i)
        out[base + (long)i * DD] = acc[r][cc][i] + bv;
    }
  }
}

// ---------------------------------------------------------------------------
extern "C" void kernel_launch(void* const* d_in, const int* in_sizes, int n_in,
                              void* d_out, int out_size, void* d_ws,
                              size_t ws_size, hipStream_t stream) {
  (void)in_sizes; (void)n_in; (void)out_size; (void)ws_size;
  const float* x      = (const float*)d_in[0];
  const float* kv_w   = (const float*)d_in[1];
  const float* kv_b   = (const float*)d_in[2];
  const float* wtq    = (const float*)d_in[3];
  const float* mix_w  = (const float*)d_in[4];
  const float* ln1_g  = (const float*)d_in[5];
  const float* ln1_b  = (const float*)d_in[6];
  const float* qkv_w  = (const float*)d_in[7];
  const float* qkv_b  = (const float*)d_in[8];
  const float* mo_w   = (const float*)d_in[9];
  const float* mo_b   = (const float*)d_in[10];
  const float* ln2_g  = (const float*)d_in[11];
  const float* ln2_b  = (const float*)d_in[12];
  const float* alphaC = (const float*)d_in[13];
  const float* out_w  = (const float*)d_in[14];
  const float* out_b  = (const float*)d_in[15];
  float* out = (float*)d_out;

  char* base = (char*)d_ws;
  size_t off = 0;
  auto alloc = [&](size_t bytes) -> void* {
    void* r = base + off;
    off += (bytes + 255) & ~(size_t)255;
    return r;
  };
  _Float16* xh      = (_Float16*)alloc((size_t)BN * DD * 2);   // reused as out1
  _Float16* kh      = (_Float16*)alloc((size_t)BN * DD * 2);
  _Float16* xvT     = (_Float16*)alloc((size_t)BB * HH * HD * NN * 2);
  _Float16* scoresT = (_Float16*)alloc((size_t)BN * DD * 2);
  float*    scores2 = (float*)   alloc((size_t)BN * DD * 4);
  _Float16* cwT     = (_Float16*)alloc((size_t)BN * DD * 2);
  _Float16* cw_mn   = (_Float16*)alloc((size_t)BB * HH * MM * NN * 2);
  float*    rs      = (float*)   alloc((size_t)BB * HH * MM * 4);
  float*    ctP     = (float*)   alloc((size_t)CTCH * BB * MM * DD * 4);
  float*    ctBMD   = (float*)   alloc((size_t)BB * MM * DD * 4);
  float*    ct_ln   = (float*)   alloc((size_t)BB * MM * DD * 4);
  float*    qkvws   = (float*)   alloc((size_t)BB * MM * 768 * 4);
  float*    attws   = (float*)   alloc((size_t)BB * HH * MM * MM * 4);
  float*    mows    = (float*)   alloc((size_t)BB * MM * DD * 4);
  _Float16* otT     = (_Float16*)alloc((size_t)BB * HH * HD * MM * 2);
  _Float16* kvwh    = (_Float16*)alloc((size_t)2 * DD * DD * 2);
  _Float16* wtqh    = (_Float16*)alloc((size_t)HH * MM * HD * 2);
  _Float16* mixh    = (_Float16*)alloc((size_t)DD * DD * 2);
  _Float16* outwh   = (_Float16*)alloc((size_t)DD * DD * 2);
  _Float16* out1    = xh;  // xh is dead after k_gemm_kv

  k_cvt<<<256, 256, 0, stream>>>(x, xh, (long)BN * DD);
  k_cvt<<<64, 256, 0, stream>>>(kv_w, kvwh, 2L * DD * DD);
  k_cvt<<<8, 256, 0, stream>>>(wtq, wtqh, (long)HH * MM * HD);
  k_cvt<<<32, 256, 0, stream>>>(mix_w, mixh, (long)DD * DD);
  k_cvt<<<32, 256, 0, stream>>>(out_w, outwh, (long)DD * DD);

  k_gemm_kv<<<dim3(BN / 32, 4), 128, 0, stream>>>(xh, kvwh, kv_b, kh, xvT);
  k_scores<<<4096, 128, 0, stream>>>(wtqh, kh, scoresT);
  k_mix<<<4096, 128, 0, stream>>>(mixh, scoresT, scores2);
  k_softmax<<<2048, 256, 0, stream>>>(scores2, cwT, cw_mn);
  k_rowsum<<<BB * HH * MM, 256, 0, stream>>>(cw_mn, rs);
  k_ct<<<BB * HH * CTCH, 128, 0, stream>>>(cw_mn, xvT, ctP);
  k_ctred<<<BB * MM * DD / 256, 256, 0, stream>>>(ctP, ctBMD);
  k_norm_ln1<<<BB * MM, 256, 0, stream>>>(ctBMD, rs, ln1_g, ln1_b, ct_ln);
  k_qkv<<<BB * MM * 768 / 256, 256, 0, stream>>>(ct_ln, qkv_w, qkv_b, qkvws);
  k_att<<<BB * HH, 32, 0, stream>>>(qkvws, attws);
  k_mo<<<BB * MM, 256, 0, stream>>>(attws, qkvws, mows);
  k_moproj_ln2<<<BB * MM, 256, 0, stream>>>(mows, mo_w, mo_b, ct_ln, alphaC,
                                            ln2_g, ln2_b, otT);
  k_bcast<<<4096, 128, 0, stream>>>(cwT, otT, out1);
  k_out<<<dim3(BN / 32, 2), 128, 0, stream>>>(out1, outwh, out_b, out);
}
